// ABHUE_16338055594437
// MI455X (gfx1250) — matrix operand order repaired and checked by reading, not compiled, not run on hardware
//
#include <hip/hip_runtime.h>
#include <hip/hip_bf16.h>
#include <math.h>

// ---------------------------------------------------------------------------
// Hierarchical LSTM (ABHUE) for MI455X / gfx1250, wave32 + WMMA bf16.
//
// ctx pass:  16 WGs x 16 sentences each (+1 WG for main-LSTM on mid sentence),
//            512 steps; per step G[16,832] = [x_t | h][16,416] @ Wcat[416,832]
//            via v_wmma_f32_16x16x32_bf16.  13 waves per WG, one 16-column
//            gate-group per wave, so i/f/g/o for the same unit land in the
//            same (VGPR,lane) slot of 4 accumulators -> LSTM cell update and
//            c-state live entirely in registers for all 512 steps.
// Weights pre-packed into per-lane WMMA-B fragment order (bf16) in workspace;
// the 2.7 MB blob stays L2-resident and streams global->VGPR->WMMA.
// Pointwise nonlinearities use branch-free native trans ops (v_tanh/v_exp/
// v_rcp) instead of libm tanhf (which emitted EXEC-mask branches per element).
// Workspace need: ~3.0 MB.
// ---------------------------------------------------------------------------

typedef __attribute__((ext_vector_type(16))) __bf16 bf16x16;
typedef __attribute__((ext_vector_type(8)))  __bf16 bf16x8;
typedef __attribute__((ext_vector_type(8)))  float  f32x8;

#define H_    200
#define S_    255
#define W_    512
#define MID_  127
#define KC    416            // padded K: x(200) + h(200) + 16 zero, = 13*32
#define KT    13             // k-tiles of 32
#define NP    832            // padded gate dim: 4 * 208
#define NT    52             // NP / 16
#define GPW   13             // 16-wide column groups per gate section (208/16)
#define FRAG  512            // bf16 elements per (nt,kt) fragment: 32 lanes * 16
#define SETSZ ((size_t)NT * KT * FRAG)      // bf16 elements per packed weight set
#define BIASP 832
#define NTHR  416            // 13 waves: one column group per wave

// Branch-free transcendentals (single hardware trans ops, no EXEC divergence).
#if __has_builtin(__builtin_amdgcn_tanhf)
__device__ __forceinline__ float htanh(float x) { return __builtin_amdgcn_tanhf(x); }
#elif __has_builtin(__builtin_amdgcn_tanh_f32)
__device__ __forceinline__ float htanh(float x) { return __builtin_amdgcn_tanh_f32(x); }
#else
__device__ __forceinline__ float htanh(float x) {
    float e = __expf(-2.0f * x);                       // v_mul + v_exp_f32
    return (1.0f - e) * __builtin_amdgcn_rcpf(1.0f + e);
}
#endif
__device__ __forceinline__ float sigm(float x) {
    return __builtin_amdgcn_rcpf(1.0f + __expf(-x));   // v_exp + v_rcp, branch-free
}

// ---------------------------------------------------------------------------
// Pack one weight set's [800,200]x2 matrices into WMMA-B fragment order:
// blob[nt][kt][lane][16 bf16]; value m of lane l is
//   Wcat[ 32*kt + 8*(l>=16) + (m&7) + 16*(m>=8) ][ 16*nt + (l&15) ]
// Wcat[k][n]: k<200 -> Wih[nrow][k]; k<400 -> Whh[nrow][k-200]; else 0,
// where n = gate*208 + j (j<200 valid, else zero pad), nrow = gate*200 + j.
// ---------------------------------------------------------------------------
__global__ void pack_params(
    const float* cWih, const float* cWhh, const float* cbi, const float* cbh,
    const float* mWih, const float* mWhh, const float* mbi, const float* mbh,
    const float* pWih, const float* pWhh, const float* pbi, const float* pbh,
    const float* qWih, const float* qWhh, const float* qbi, const float* qbh,
    unsigned short* blob, float* biasp)
{
    int b   = blockIdx.x;            // set*NT + nt
    int set = b / NT, nt = b % NT;
    const float *Wih, *Whh, *bi, *bh;
    if      (set == 0) { Wih = cWih; Whh = cWhh; bi = cbi; bh = cbh; }
    else if (set == 1) { Wih = mWih; Whh = mWhh; bi = mbi; bh = mbh; }
    else if (set == 2) { Wih = pWih; Whh = pWhh; bi = pbi; bh = pbh; }
    else               { Wih = qWih; Whh = qWhh; bi = qbi; bh = qbh; }

    unsigned short* out = blob + (size_t)set * SETSZ;
    for (int kt = 0; kt < KT; ++kt) {
        for (int vi = threadIdx.x; vi < FRAG; vi += blockDim.x) {
            int lane = vi >> 4, m = vi & 15;
            int k = 32 * kt + ((lane >= 16) ? 8 : 0) + (m & 7) + ((m >= 8) ? 16 : 0);
            int n = 16 * nt + (lane & 15);
            int gate = n / 208, j = n % 208;
            float v = 0.0f;
            if (j < 200 && k < 400) {
                int r = gate * 200 + j;
                v = (k < 200) ? Wih[r * 200 + k] : Whh[r * 200 + (k - 200)];
            }
            union { __bf16 b; unsigned short u; } cv;
            cv.b = (__bf16)v;
            out[((size_t)(nt * KT + kt) * 32 + lane) * 16 + m] = cv.u;
        }
    }
    if (nt == 0) {   // padded combined bias
        for (int idx = threadIdx.x; idx < NP; idx += blockDim.x) {
            int gate = idx / 208, j = idx % 208;
            biasp[set * BIASP + idx] =
                (j < 200) ? (bi[gate * 200 + j] + bh[gate * 200 + j]) : 0.0f;
        }
    }
}

// One wave, one 16-column group g: 4 gate accumulators, 13 k-steps of WMMA,
// then the LSTM cell pointwise update entirely in registers.
__device__ __forceinline__ void lstm_group_step(
    int g, int lane, __bf16 (*xh)[KC], const __bf16* wb,
    const float bias4[4], float c[8])
{
    f32x8 acc0, acc1, acc2, acc3;
#pragma unroll
    for (int e = 0; e < 8; ++e) {
        acc0[e] = bias4[0]; acc1[e] = bias4[1];
        acc2[e] = bias4[2]; acc3[e] = bias4[3];
    }
    const int arow = lane & 15;            // M (A layout: both halves share M=lane&15)
    const int koff = (lane >> 4) * 8;      // lanes 0-15: K {0..7,16..23}; 16-31: {8..15,24..31}

    // per-gate fragment base pointers for this column group (lane-resolved)
    const __bf16* bp0 = wb + ((size_t)((0 * GPW + g) * KT) * 32 + lane) * 16;
    const __bf16* bp1 = wb + ((size_t)((1 * GPW + g) * KT) * 32 + lane) * 16;
    const __bf16* bp2 = wb + ((size_t)((2 * GPW + g) * KT) * 32 + lane) * 16;
    const __bf16* bp3 = wb + ((size_t)((3 * GPW + g) * KT) * 32 + lane) * 16;

#pragma unroll 1
    for (int kt = 0; kt < KT; ++kt) {
        union { bf16x16 v; bf16x8 h[2]; } af;
        af.h[0] = *(const bf16x8*)&xh[arow][32 * kt + koff];
        af.h[1] = *(const bf16x8*)&xh[arow][32 * kt + koff + 16];
        const size_t ko = (size_t)kt * FRAG;
        bf16x16 b0 = *(const bf16x16*)(bp0 + ko);
        bf16x16 b1 = *(const bf16x16*)(bp1 + ko);
        bf16x16 b2 = *(const bf16x16*)(bp2 + ko);
        bf16x16 b3 = *(const bf16x16*)(bp3 + ko);
        acc0 = __builtin_amdgcn_wmma_f32_16x16x32_bf16(false, af.v, false, b0, (short)0, acc0, false, false);
        acc1 = __builtin_amdgcn_wmma_f32_16x16x32_bf16(false, af.v, false, b1, (short)0, acc1, false, false);
        acc2 = __builtin_amdgcn_wmma_f32_16x16x32_bf16(false, af.v, false, b2, (short)0, acc2, false, false);
        acc3 = __builtin_amdgcn_wmma_f32_16x16x32_bf16(false, af.v, false, b3, (short)0, acc3, false, false);
    }
    // C/D layout: element e, lane l -> sentence = e + 8*(l>=16), unit-col = l&15
#pragma unroll
    for (int e = 0; e < 8; ++e) {
        float cn = sigm(acc1[e]) * c[e] + sigm(acc0[e]) * htanh(acc2[e]);
        float h  = sigm(acc3[e]) * htanh(cn);
        c[e] = cn;
        int rowE = e + ((lane >> 4) << 3);
        xh[rowE][200 + 16 * g + (lane & 15)] = (__bf16)h;
    }
}

// mode 0: blocks 0..15 ctx (16 sentences each), block 16 main (row 0 = mid)
// mode 1: block 0 prev over sent_emb[0..mid], block 1 post (reversed tail)
__global__ void __launch_bounds__(NTHR) lstm_tile(
    const float* __restrict__ utter, const float* __restrict__ h0_sent,
    const float* __restrict__ h0_prev, const float* __restrict__ h0_post,
    const unsigned short* __restrict__ blobu, const float* __restrict__ biasp,
    float* __restrict__ sent_emb, float* __restrict__ feat, int mode)
{
    __shared__ __bf16 xh[16][KC];
    const int tid = threadIdx.x;
    const int g = tid >> 5, lane = tid & 31;   // 13 waves -> group g = wave id
    const int bid = blockIdx.x;

    int set, T, s0 = 0;
    bool isCtx = false, isMain = false, isPrev = false, isPost = false;
    if (mode == 0) {
        if (bid < 16) { isCtx = true; set = 0; s0 = bid * 16; T = W_; }
        else          { isMain = true; set = 1; T = W_; }
    } else {
        if (bid == 0) { isPrev = true; set = 2; }
        else          { isPost = true; set = 3; }
        T = MID_ + 1;
    }
    const __bf16* wb = reinterpret_cast<const __bf16*>(blobu) + (size_t)set * SETSZ;
    const float*  bp = biasp + set * BIASP;

    // init h section (cols 200..399) from h0, pad cols 400..415 = 0
    for (int i = tid; i < 16 * (KC - 200); i += NTHR) {
        int row = i / (KC - 200);
        int col = 200 + i % (KC - 200);
        float hv = 0.0f;
        if (col < 400) {
            int j = col - 200;
            if (isCtx)       { int s = s0 + row; if (s < S_) hv = h0_sent[s * H_ + j]; }
            else if (isMain) { if (row == 0) hv = h0_sent[MID_ * H_ + j]; }
            else if (isPrev) { if (row == 0) hv = h0_prev[j]; }
            else             { if (row == 0) hv = h0_post[j]; }
        }
        xh[row][col] = (__bf16)hv;
    }

    // per-wave state for this wave's column group
    float bias4[4], c[8];
#pragma unroll
    for (int q = 0; q < 4; ++q)
        bias4[q] = bp[q * 208 + 16 * g + (lane & 15)];
#pragma unroll
    for (int e = 0; e < 8; ++e) c[e] = 0.0f;
    __syncthreads();

#pragma unroll 1
    for (int t = 0; t < T; ++t) {
        // stage x_t (f32 -> bf16) into xh[:, 0..199]
        for (int i = tid; i < 16 * 200; i += NTHR) {
            int row = i / 200, col = i % 200;
            float xv = 0.0f;
            if (isCtx)       { int s = s0 + row; if (s < S_) xv = utter[((size_t)s * W_ + t) * H_ + col]; }
            else if (isMain) { if (row == 0) xv = utter[((size_t)MID_ * W_ + t) * H_ + col]; }
            else if (isPrev) { if (row == 0) xv = sent_emb[t * H_ + col]; }
            else             { if (row == 0) xv = sent_emb[(S_ - 1 - t) * H_ + col]; }
            xh[row][col] = (__bf16)xv;
        }
        __syncthreads();
        lstm_group_step(g, lane, xh, wb, bias4, c);   // all 13 waves, EXEC all-ones
        __syncthreads();
    }

    // emit final hidden states
    for (int i = tid; i < 16 * 200; i += NTHR) {
        int row = i / 200, col = i % 200;
        float h = (float)xh[row][200 + col];
        if (isCtx)       { int s = s0 + row; if (s < S_ && s != MID_) sent_emb[s * H_ + col] = h; }
        else if (isMain) { if (row == 0) sent_emb[MID_ * H_ + col] = h; }
        else if (isPrev) { if (row == 0) feat[col] = h; }
        else             { if (row == 0) feat[200 + col] = h; }
    }
}

__global__ void fc_kernel(const float* __restrict__ feat,
                          const float* __restrict__ fcW,
                          const float* __restrict__ fcb,
                          float* __restrict__ out)
{
    int j = blockIdx.x * blockDim.x + threadIdx.x;
    if (j < H_) {
        float s = fcb[j];
        for (int k = 0; k < 2 * H_; ++k) s += feat[k] * fcW[j * 2 * H_ + k];
        out[j] = s;
    }
}

extern "C" void kernel_launch(void* const* d_in, const int* in_sizes, int n_in,
                              void* d_out, int out_size, void* d_ws, size_t ws_size,
                              hipStream_t stream)
{
    (void)in_sizes; (void)n_in; (void)out_size; (void)ws_size;
    const float* utter   = (const float*)d_in[0];
    const float* h0_sent = (const float*)d_in[1];
    const float* h0_prev = (const float*)d_in[2];
    const float* h0_post = (const float*)d_in[3];
    const float* W[16];
    for (int i = 0; i < 16; ++i) W[i] = (const float*)d_in[4 + i];
    const float* fcW = (const float*)d_in[20];
    const float* fcb = (const float*)d_in[21];
    float* out = (float*)d_out;

    // workspace layout (~3.0 MB total)
    char* ws = (char*)d_ws;
    size_t off = 0;
    unsigned short* blob = (unsigned short*)(ws + off);
    off += 4 * SETSZ * sizeof(unsigned short);  off = (off + 255) & ~(size_t)255;
    float* biasp = (float*)(ws + off);
    off += 4 * BIASP * sizeof(float);           off = (off + 255) & ~(size_t)255;
    float* sent_emb = (float*)(ws + off);
    off += (size_t)S_ * H_ * sizeof(float);     off = (off + 255) & ~(size_t)255;
    float* feat = (float*)(ws + off);

    pack_params<<<4 * NT, 256, 0, stream>>>(
        W[0], W[1], W[2], W[3],  W[4], W[5], W[6], W[7],
        W[8], W[9], W[10], W[11], W[12], W[13], W[14], W[15],
        blob, biasp);
    lstm_tile<<<17, NTHR, 0, stream>>>(utter, h0_sent, h0_prev, h0_post,
                                       blob, biasp, sent_emb, feat, 0);
    lstm_tile<<<2, NTHR, 0, stream>>>(utter, h0_sent, h0_prev, h0_post,
                                      blob, biasp, sent_emb, feat, 1);
    fc_kernel<<<1, 256, 0, stream>>>(feat, fcW, fcb, out);
}
